// Memory_41317585388075
// MI455X (gfx1250) — matrix operand order repaired and checked by reading, not compiled, not run on hardware
//
#include <hip/hip_runtime.h>
#include <hip/hip_bf16.h>
#include <math.h>

// ---------- types for CDNA5 WMMA ----------
typedef __attribute__((ext_vector_type(16))) __bf16 v16bf;
typedef __attribute__((ext_vector_type(8)))  __bf16 v8bf;
typedef __attribute__((ext_vector_type(4)))  __bf16 v4bf;
typedef __attribute__((ext_vector_type(8)))  float  v8f;
typedef __attribute__((ext_vector_type(4)))  float  v4f;

// ---------- problem constants ----------
#define NMEM   2000
#define CCH    512
#define HH     7
#define WW7    7
#define BB     4
#define SP     49                 // H*W
#define KDIM   (CCH * SP)         // 25088
#define MROWS  (BB * SP)          // 196
#define MPAD   208                // 13 * 16
#define MT     13                 // M tiles of 16
#define NPAD   2000               // 125 * 16 exactly
#define KSPLIT 8
#define KSLICE (KDIM / KSPLIT)    // 3136 = 49 chunks of 64
#define KCHUNK 64
#define PI2F   1.570795f          // 3.14159 / 2 (matches reference literal)
#define CTILE  16
#define NCHUNKS 10
#define NPERCH  (NMEM / NCHUNKS)  // 200

// =====================================================================
// Kernel 1: L2 norms of x rows (B) and mem rows (N)
// =====================================================================
__global__ __launch_bounds__(256) void norms_kernel(const float* __restrict__ x,
                                                    const float* __restrict__ mem,
                                                    float* __restrict__ mn,
                                                    float* __restrict__ xn) {
  __shared__ float red[256];
  const int row = blockIdx.x;
  const float* src = (row < NMEM) ? (mem + (size_t)row * KDIM)
                                  : (x + (size_t)(row - NMEM) * KDIM);
  float s = 0.f;
  for (int i = threadIdx.x; i < KDIM; i += 256) { float v = src[i]; s += v * v; }
  red[threadIdx.x] = s;
  __syncthreads();
  for (int off = 128; off > 0; off >>= 1) {
    if (threadIdx.x < off) red[threadIdx.x] += red[threadIdx.x + off];
    __syncthreads();
  }
  if (threadIdx.x == 0) {
    float nv = sqrtf(red[0]);
    if (row < NMEM) mn[row] = nv; else xn[row - NMEM] = nv;
  }
}

// =====================================================================
// Kernel 2: build shifted-x operand Xshift[(b,s)][(c,p)] in bf16
//   Xshift[b*49+s][c*49+p] = x[b,c,(h-dy)%7,(w-dx)%7],  rows 196..207 = 0
// =====================================================================
__global__ __launch_bounds__(256) void xshift_kernel(const float* __restrict__ x,
                                                     __bf16* __restrict__ xs) {
  const int e = blockIdx.x * 256 + threadIdx.x;
  if (e >= MPAD * KDIM) return;
  const int row = e / KDIM;
  const int k   = e - row * KDIM;
  float v = 0.f;
  if (row < MROWS) {
    const int b = row / SP, s = row - b * SP;
    const int dy = s / WW7, dx = s - dy * WW7;
    const int c = k / SP, p = k - c * SP;
    const int h = p / WW7, w = p - h * WW7;
    int hs = h - dy; if (hs < 0) hs += HH;
    int ws = w - dx; if (ws < 0) ws += WW7;
    v = x[(size_t)b * KDIM + c * SP + hs * WW7 + ws];
  }
  xs[e] = (__bf16)v;
}

// =====================================================================
// Kernel 3: correlation GEMM  corr[m,n] = sum_k Xshift[m,k] * mem[n,k]
//   M = 208 (13 WMMA tiles), N tile of 16 per block, K split across gridDim.y.
//   256 threads = 8 waves; wave w owns M-tiles {w, w+8} (13 used).
//   B staging: branch-free, 4 fp32 per thread per 64-K chunk:
//     1x global_load_b128 -> 2x v_cvt_pk_bf16_f32 -> 1x ds_store_b64,
//   software-pipelined one chunk ahead. 4 WMMAs per wave per barrier pair.
// =====================================================================
__global__ __launch_bounds__(256) void corr_gemm_kernel(const __bf16* __restrict__ xs,
                                                        const float* __restrict__ mem,
                                                        float* __restrict__ corr) {
  // two 32-K sub-blocks, each [32 lanes][16 bf16] in B register layout
  __shared__ alignas(32) __bf16 ldsB[2 * 32 * 16];
  const int tid   = threadIdx.x;
  const int lane  = tid & 31;
  const int wv    = tid >> 5;                   // 0..7
  const int n0    = blockIdx.x * 16;
  const int kbeg  = blockIdx.y * KSLICE;
  const int kend  = kbeg + KSLICE;
  const int ml    = lane & 15;                  // row-in-tile for A, col for B/D
  const int half  = lane >> 4;

  const int mt0 = wv;                           // always < 13
  const int mt1 = wv + 8;                       // valid for wv < 5
  const bool has2 = (mt1 < MT);                 // wave-uniform
  const __bf16* arow0 = xs + (size_t)(mt0 * 16 + ml) * KDIM;
  const __bf16* arow1 = xs + (size_t)((has2 ? mt1 : mt0) * 16 + ml) * KDIM;

  // ----- branch-free fill slot: thread i handles 4 consecutive k -----
  const int fni = tid >> 4;                     // n index 0..15
  const int fkk = (tid & 15) * 4;               // k offset 0..60, mult of 4
  const float* gbase = mem + (size_t)(n0 + fni) * KDIM + fkk;  // + kb at use
  const int fOff = (fkk >> 5) * 512 + ((((fkk & 8) != 0) ? 16 : 0) + fni) * 16 + (fkk & 7) + ((fkk & 16) >> 1);

  v8f acc0 = {};
  v8f acc1 = {};

  // prologue: stage chunk 0
  v4f cur = *(const v4f*)(gbase + kbeg);

  for (int kb = kbeg; kb < kend; kb += KCHUNK) {
    __syncthreads();                   // previous chunk's LDS reads done
    {
      v4bf bv;
      bv[0] = (__bf16)cur[0]; bv[1] = (__bf16)cur[1];
      bv[2] = (__bf16)cur[2]; bv[3] = (__bf16)cur[3];
      *(v4bf*)(&ldsB[fOff]) = bv;      // ds_store_b64
    }
    __syncthreads();                   // tile published

    // kick off next chunk's global load (hidden behind WMMAs below)
    const int kn = kb + KCHUNK;
    if (kn < kend) {
      cur = *(const v4f*)(gbase + kn);
      __builtin_prefetch(gbase + kn + KCHUNK, 0, 0);  // global_prefetch_b8
    }

    const v16bf b0 = *(const v16bf*)(&ldsB[lane * 16]);
    const v16bf b1 = *(const v16bf*)(&ldsB[512 + lane * 16]);

    {
      const __bf16* ap = arow0 + kb;
      v8bf alo0 = *(const v8bf*)(ap + half * 8);
      v8bf ahi0 = *(const v8bf*)(ap + 16 + half * 8);
      v8bf alo1 = *(const v8bf*)(ap + 32 + half * 8);
      v8bf ahi1 = *(const v8bf*)(ap + 48 + half * 8);
      v16bf a0 = __builtin_shufflevector(alo0, ahi0,
          0, 1, 2, 3, 4, 5, 6, 7, 8, 9, 10, 11, 12, 13, 14, 15);
      v16bf a1 = __builtin_shufflevector(alo1, ahi1,
          0, 1, 2, 3, 4, 5, 6, 7, 8, 9, 10, 11, 12, 13, 14, 15);
      acc0 = __builtin_amdgcn_wmma_f32_16x16x32_bf16(
          false, a0, false, b0, (short)0, acc0, false, false);
      acc0 = __builtin_amdgcn_wmma_f32_16x16x32_bf16(
          false, a1, false, b1, (short)0, acc0, false, false);
    }
    if (has2) {   // wave-uniform
      const __bf16* ap = arow1 + kb;
      v8bf alo0 = *(const v8bf*)(ap + half * 8);
      v8bf ahi0 = *(const v8bf*)(ap + 16 + half * 8);
      v8bf alo1 = *(const v8bf*)(ap + 32 + half * 8);
      v8bf ahi1 = *(const v8bf*)(ap + 48 + half * 8);
      v16bf a0 = __builtin_shufflevector(alo0, ahi0,
          0, 1, 2, 3, 4, 5, 6, 7, 8, 9, 10, 11, 12, 13, 14, 15);
      v16bf a1 = __builtin_shufflevector(alo1, ahi1,
          0, 1, 2, 3, 4, 5, 6, 7, 8, 9, 10, 11, 12, 13, 14, 15);
      acc1 = __builtin_amdgcn_wmma_f32_16x16x32_bf16(
          false, a0, false, b0, (short)0, acc1, false, false);
      acc1 = __builtin_amdgcn_wmma_f32_16x16x32_bf16(
          false, a1, false, b1, (short)0, acc1, false, false);
    }
  }

  // store D: lane 0-15 -> rows r+0, lane 16-31 -> rows r+8; col = n0 + (lane&15)
  float* cbase = corr + (size_t)blockIdx.y * MPAD * NPAD;
  const int ncol = n0 + ml;
  const int mb0 = mt0 * 16 + half * 8;
#pragma unroll
  for (int rr = 0; rr < 8; ++rr)
    cbase[(size_t)(mb0 + rr) * NPAD + ncol] = acc0[rr];
  if (has2) {
    const int mb1 = mt1 * 16 + half * 8;
#pragma unroll
    for (int rr = 0; rr < 8; ++rr)
      cbase[(size_t)(mb1 + rr) * NPAD + ncol] = acc1[rr];
  }
}

// =====================================================================
// Kernel 4: per (b,n): sum K-partials, argmax over 49 shifts, tan logit
// =====================================================================
__global__ __launch_bounds__(256) void argmax_kernel(const float* __restrict__ corr,
                                                     const float* __restrict__ xn,
                                                     const float* __restrict__ mn,
                                                     float* __restrict__ logit,
                                                     int* __restrict__ trans) {
  const int g = blockIdx.x * 256 + threadIdx.x;
  if (g >= BB * NMEM) return;
  const int b = g / NMEM, n = g - b * NMEM;
  float best = -3.0e38f; int bs = 0;
  for (int s = 0; s < SP; ++s) {
    float v = 0.f;
#pragma unroll
    for (int ks = 0; ks < KSPLIT; ++ks)
      v += corr[((size_t)ks * MPAD + b * SP + s) * NPAD + n];
    if (v > best) { best = v; bs = s; }
  }
  const float sim = best / (xn[b] * mn[n]);
  logit[g] = tanf(sim * PI2F);    // * RR (== 1.0)
  trans[g] = bs;
}

// =====================================================================
// Kernel 5: softmax over N per batch
// =====================================================================
__global__ __launch_bounds__(256) void softmax_kernel(const float* __restrict__ logit,
                                                      float* __restrict__ w) {
  __shared__ float red[256];
  const int b = blockIdx.x;
  const float* l = logit + b * NMEM;
  float m = -3.0e38f;
  for (int i = threadIdx.x; i < NMEM; i += 256) m = fmaxf(m, l[i]);
  red[threadIdx.x] = m; __syncthreads();
  for (int off = 128; off > 0; off >>= 1) {
    if (threadIdx.x < off) red[threadIdx.x] = fmaxf(red[threadIdx.x], red[threadIdx.x + off]);
    __syncthreads();
  }
  m = red[0]; __syncthreads();
  float s = 0.f;
  for (int i = threadIdx.x; i < NMEM; i += 256) s += expf(l[i] - m);
  red[threadIdx.x] = s; __syncthreads();
  for (int off = 128; off > 0; off >>= 1) {
    if (threadIdx.x < off) red[threadIdx.x] += red[threadIdx.x + off];
    __syncthreads();
  }
  const float inv = 1.f / red[0];
  for (int i = threadIdx.x; i < NMEM; i += 256) w[b * NMEM + i] = expf(l[i] - m) * inv;
}

// =====================================================================
// Kernel 6: roll-aligned weighted sum (partials over n-chunks)
//   out[b,c,h,w] += w[b,n] * mem[n,c,(h+dy)%7,(w+dx)%7]
//   grid: (32 channel tiles of 16) x (10 n-chunks of 200); block 256 (196 active)
// =====================================================================
__global__ __launch_bounds__(256) void wsum_kernel(const float* __restrict__ mem,
                                                   const float* __restrict__ w,
                                                   const int* __restrict__ trans,
                                                   float* __restrict__ part) {
  __shared__ float tile[CTILE * SP];   // 784 floats
  __shared__ float wsh[BB];
  __shared__ int   ssh[BB];
  const int ct = blockIdx.x;           // channel tile
  const int ch = blockIdx.y;           // n chunk
  const int t  = threadIdx.x;
  const bool act = t < 196;
  int p = 0, cp = 0, h = 0, ww = 0;
  if (act) { p = t % SP; cp = t / SP; h = p / WW7; ww = p - h * WW7; }
  float acc[4][BB];
  for (int i = 0; i < 4; ++i) for (int b = 0; b < BB; ++b) acc[i][b] = 0.f;

  const size_t cbase = (size_t)ct * CTILE * SP;
  const int nbeg = ch * NPERCH;
  for (int n = nbeg; n < nbeg + NPERCH; ++n) {
    __syncthreads();
    if (act) {
      const float* src = mem + (size_t)n * KDIM + cbase;
      tile[t]       = src[t];
      tile[t + 196] = src[t + 196];
      tile[t + 392] = src[t + 392];
      tile[t + 588] = src[t + 588];
    }
    if (t < BB)               wsh[t] = w[t * NMEM + n];
    else if (t < 2 * BB)      ssh[t - BB] = trans[(t - BB) * NMEM + n];
    __syncthreads();
    if (act) {
      for (int b = 0; b < BB; ++b) {
        const int s  = ssh[b];
        const int dy = s / WW7, dx = s - dy * WW7;
        int hh = h + dy;  if (hh >= HH)  hh -= HH;
        int wp = ww + dx; if (wp >= WW7) wp -= WW7;
        const int pp = hh * WW7 + wp;
        const float wt = wsh[b];
        acc[0][b] += wt * tile[(cp * 4 + 0) * SP + pp];
        acc[1][b] += wt * tile[(cp * 4 + 1) * SP + pp];
        acc[2][b] += wt * tile[(cp * 4 + 2) * SP + pp];
        acc[3][b] += wt * tile[(cp * 4 + 3) * SP + pp];
      }
    }
  }
  __syncthreads();
  if (act) {
    for (int cl = 0; cl < 4; ++cl) {
      const int c = ct * CTILE + cp * 4 + cl;
      for (int b = 0; b < BB; ++b)
        part[(size_t)ch * (BB * KDIM) + (size_t)b * KDIM + c * SP + p] = acc[cl][b];
    }
  }
}

// =====================================================================
// Kernel 7: reduce n-chunk partials -> out
// =====================================================================
__global__ __launch_bounds__(256) void reduce_kernel(const float* __restrict__ part,
                                                     float* __restrict__ out) {
  const int i = blockIdx.x * 256 + threadIdx.x;
  if (i >= BB * KDIM) return;
  float s = 0.f;
  for (int c = 0; c < NCHUNKS; ++c) s += part[(size_t)c * (BB * KDIM) + i];
  out[i] = s;
}

// =====================================================================
static inline size_t alignup(size_t x) { return (x + 255) & ~(size_t)255; }

extern "C" void kernel_launch(void* const* d_in, const int* in_sizes, int n_in,
                              void* d_out, int out_size, void* d_ws, size_t ws_size,
                              hipStream_t stream) {
  (void)in_sizes; (void)n_in; (void)out_size; (void)ws_size;
  const float* x   = (const float*)d_in[0];   // key_input [4,512,7,7]
  const float* mem = (const float*)d_in[1];   // memory    [2000,512,7,7]
  float* out = (float*)d_out;

  char* ws = (char*)d_ws;
  size_t off = 0;
  __bf16* xs   = (__bf16*)(ws + off); off = alignup(off + (size_t)MPAD * KDIM * sizeof(__bf16));
  float*  corr = (float*)(ws + off);  off = alignup(off + (size_t)KSPLIT * MPAD * NPAD * sizeof(float));
  float*  mn   = (float*)(ws + off);  off = alignup(off + (size_t)NMEM * sizeof(float));
  float*  xn   = (float*)(ws + off);  off = alignup(off + (size_t)BB * sizeof(float));
  float*  lg   = (float*)(ws + off);  off = alignup(off + (size_t)BB * NMEM * sizeof(float));
  int*    tr   = (int*)(ws + off);    off = alignup(off + (size_t)BB * NMEM * sizeof(int));
  float*  wbn  = (float*)(ws + off);  off = alignup(off + (size_t)BB * NMEM * sizeof(float));
  float*  part = (float*)(ws + off);  off = alignup(off + (size_t)NCHUNKS * BB * KDIM * sizeof(float));

  norms_kernel<<<NMEM + BB, 256, 0, stream>>>(x, mem, mn, xn);

  const int xsElems = MPAD * KDIM;
  xshift_kernel<<<(xsElems + 255) / 256, 256, 0, stream>>>(x, xs);

  corr_gemm_kernel<<<dim3(NPAD / 16, KSPLIT), 256, 0, stream>>>(xs, mem, corr);

  argmax_kernel<<<(BB * NMEM + 255) / 256, 256, 0, stream>>>(corr, xn, mn, lg, tr);

  softmax_kernel<<<BB, 256, 0, stream>>>(lg, wbn);

  wsum_kernel<<<dim3(CCH / CTILE, NCHUNKS), 256, 0, stream>>>(mem, wbn, tr, part);

  reduce_kernel<<<(BB * KDIM + 255) / 256, 256, 0, stream>>>(part, out);
}
